// ReactiveModel_77068893160225
// MI455X (gfx1250) — matrix operand order repaired and checked by reading, not compiled, not run on hardware
//
#include <hip/hip_runtime.h>

// ---------------------------------------------------------------------------
// CDNA5 (gfx1250) persistent GRU + entropy-gated attention.
//   kernel 1: zero per-step sync slots
//   kernel 2: pack weights into f16 per-lane WMMA B-fragments (ws)
//   kernel 3: K/V precompute  (WMMA f16)  -> 64MB f16, L2-resident
//   kernel 4: persistent GRU loop, 16 blocks x 256 thr, WMMA gates/logits/q,
//             grid-wide entropy sync per step, VALU attention from L2 K/V
// ---------------------------------------------------------------------------

typedef __attribute__((ext_vector_type(16))) _Float16 v16h;
typedef __attribute__((ext_vector_type(8)))  float    v8f;

#define HD   64
#define VOC  64
#define BT   64          // batch rows per block
#define NT   256         // threads per block (8 waves of 32)
#define ALD  192         // leading dim of fused A = [x | retrieved | h]
#define ENT_THRESH 2.5f
#define ATT_SCALE  0.125f  // 1/sqrt(64)

// dynamic-LDS layout (bytes)
#define A_OFF    0        // f16 [64][192]  fused A            24576
#define G_OFF    24576    // f16 [64][192]  gate preacts       24576
#define GHN_OFF  49152    // f16 [64][64]   h_n part            8192
#define HN_OFF   57344    // f16 [64][64]   h_new               8192
#define U_OFF    65536    // f32 logits[64][64] (16384) overlaid by att f32
#define Q_SUB    16384    //   + f16 q[64][64]   (8192)
#define SC_OFF   90112    // scratch: sEnt, sCond
#define SMEM_BYTES 90368

static __device__ __forceinline__ v8f wmma16(v16h a, v16h b, v8f c) {
  return __builtin_amdgcn_wmma_f32_16x16x32_f16(false, a, false, b,
                                                (short)0, c, false, false);
}

// A fragment (16x32 f16) from row-major source, contiguous k order:
// lane l holds row m = l&15; halves 0..15 -> K = kk+0..15
// with kk = (l>=16)?16:0 (matches packed B).  Contiguous 16-half run.
static __device__ __forceinline__ v16h load_a_f16(const _Float16* p, int ld, int lane) {
  const int m  = lane & 15;
  const int kk = (lane >> 4) << 4;
  const _Float16* r = p + m * ld + kk;
  v16h a;
#pragma unroll
  for (int i = 0; i < 16; ++i) { a[i] = r[i]; }
  return a;
}

// B fragment: pre-packed per-lane (tile = 32 lanes x 16 halves = 1KB), one v16h load.
static __device__ __forceinline__ v16h load_b_packed(const _Float16* base, int tile, int lane) {
  return *(const v16h*)(base + ((size_t)tile << 9) + (lane << 4));
}

// ---------------------------------------------------------------------------
__global__ void init_sync_kernel(float* entSum, unsigned int* entCnt, int L) {
  int i = blockIdx.x * blockDim.x + threadIdx.x;
  if (i < L) { entSum[i] = 0.f; entCnt[i] = 0u; }
}

// ---------------------------------------------------------------------------
// pack a 64x64 weight W (row-major [out,in]) into B fragments: B[k][n] = W[n*64+k]
static __device__ void pack64(const float* W, _Float16* dst, int tid, int nth) {
  for (int i = tid; i < 8 * 512; i += nth) {
    int tile = i >> 9, rem = i & 511;
    int lane = rem >> 4, e = rem & 15;
    int kt = tile >> 2, nt = tile & 3;
    int k = kt * 32 + ((lane >> 4) << 4) + e;
    int n = nt * 16 + (lane & 15);
    dst[i] = (_Float16)W[n * 64 + k];
  }
}

__global__ void pack_kernel(const float* w_ih, const float* w_hh,
                            const float* b_ih, const float* b_hh,
                            const float* wq, const float* wk, const float* wv,
                            const float* head_w, const float* embed,
                            _Float16* Gw, _Float16* Whn, _Float16* Hw,
                            _Float16* Qw, _Float16* Kw, _Float16* Vw,
                            float* bsum, float* bhhn, _Float16* embed_h) {
  int tid = blockIdx.x * blockDim.x + threadIdx.x;
  int nth = gridDim.x * blockDim.x;
  // fused gate weights: B[192][192], k<128 -> w_ih[n][k], else w_hh[n][k-128]
  for (int i = tid; i < 72 * 512; i += nth) {
    int tile = i >> 9, rem = i & 511;
    int lane = rem >> 4, e = rem & 15;
    int kt = tile / 12, nt = tile % 12;
    int k = kt * 32 + ((lane >> 4) << 4) + e;
    int n = nt * 16 + (lane & 15);
    float v = (k < 128) ? w_ih[n * 128 + k] : w_hh[n * 64 + (k - 128)];
    Gw[i] = (_Float16)v;
  }
  pack64(w_hh + 128 * 64, Whn, tid, nth);  // h_n slice of w_hh
  pack64(head_w, Hw, tid, nth);
  pack64(wq, Qw, tid, nth);
  pack64(wk, Kw, tid, nth);
  pack64(wv, Vw, tid, nth);
  for (int i = tid; i < 192;     i += nth) bsum[i]    = b_ih[i] + b_hh[i];
  for (int i = tid; i < 64;      i += nth) bhhn[i]    = b_hh[128 + i];
  for (int i = tid; i < 64 * 64; i += nth) embed_h[i] = (_Float16)embed[i];
}

// ---------------------------------------------------------------------------
// K = memory @ wk^T + bk ; V = memory @ wv^T + bv ; f16 out (L2-resident)
__global__ void __launch_bounds__(NT)
kv_kernel(const float* __restrict__ memory,
          const _Float16* __restrict__ Kw, const _Float16* __restrict__ Vw,
          const float* __restrict__ bk, const float* __restrict__ bvp,
          _Float16* __restrict__ K16, _Float16* __restrict__ V16) {
  __shared__ _Float16 sM[64 * 64];
  const int tid = threadIdx.x, lane = tid & 31, wid = tid >> 5;
  const size_t g0 = (size_t)blockIdx.x * 64;   // 64 (b,m) rows per block
  for (int i = tid; i < 64 * 64; i += NT) sM[i] = (_Float16)memory[g0 * 64 + i];
  __syncthreads();
#pragma unroll
  for (int a = 0; a < 4; ++a) {
    int id = wid * 4 + a;            // 32 output tiles: 16 for K, 16 for V
    int mat = id >> 4;
    int tt = id & 15;
    int mt = tt >> 2, nt = tt & 3;
    int col = nt * 16 + (lane & 15);
    float bb = (mat ? bvp : bk)[col];
    const _Float16* Bw = mat ? Vw : Kw;
    v8f c;
#pragma unroll
    for (int r = 0; r < 8; ++r) c[r] = bb;
#pragma unroll
    for (int kt = 0; kt < 2; ++kt) {
      v16h av = load_a_f16(sM + mt * 16 * 64 + kt * 32, 64, lane);
      v16h bw = load_b_packed(Bw, kt * 4 + nt, lane);
      c = wmma16(av, bw, c);
    }
    _Float16* outp = mat ? V16 : K16;
    int rbase = mt * 16 + ((lane >> 4) << 3);
#pragma unroll
    for (int r = 0; r < 8; ++r)
      outp[(g0 + rbase + r) * 64 + col] = (_Float16)c[r];
  }
}

// ---------------------------------------------------------------------------
__global__ void __launch_bounds__(NT)
gru_kernel(const int* __restrict__ seq,
           const _Float16* __restrict__ K16, const _Float16* __restrict__ V16,
           const _Float16* __restrict__ Gw, const _Float16* __restrict__ Whn,
           const _Float16* __restrict__ Hw, const _Float16* __restrict__ Qw,
           const float* __restrict__ bsum, const float* __restrict__ bhhn,
           const _Float16* __restrict__ embed_h,
           const float* __restrict__ bq, const float* __restrict__ head_b,
           float* entSum, unsigned int* entCnt,
           float* __restrict__ out, int B, int L, int M) {
  extern __shared__ char smem[];
  _Float16* sA   = (_Float16*)(smem + A_OFF);
  _Float16* sG   = (_Float16*)(smem + G_OFF);
  _Float16* sGhn = (_Float16*)(smem + GHN_OFF);
  _Float16* sHn  = (_Float16*)(smem + HN_OFF);
  float*    sLg  = (float*)(smem + U_OFF);
  float*    sAtt = (float*)(smem + U_OFF);           // overlays logits
  _Float16* sQ   = (_Float16*)(smem + U_OFF + Q_SUB);
  float*    sEnt = (float*)(smem + SC_OFF);
  float*    sCond= (float*)(smem + SC_OFF + 4);

  const int tid = threadIdx.x, lane = tid & 31, wid = tid >> 5;
  const int r0 = blockIdx.x * BT;
  const int nblk = gridDim.x;
  int rc = 0;

  for (int i = tid; i < BT * 128; i += NT) {       // retrieved=0, h=0
    int row = i >> 7, c = i & 127;
    sA[row * ALD + 64 + c] = (_Float16)0.f;
  }
  __syncthreads();

  for (int t = 0; t < L; ++t) {
    // ---- phase 1: x_t = embed[seq[:,t]] into A cols [0,64)
    {
      int row = tid >> 2, part = tid & 3;
      int token = seq[(size_t)(r0 + row) * L + t];
      const _Float16* er = embed_h + token * 64 + part * 16;
      _Float16* dst = sA + row * ALD + part * 16;
#pragma unroll
      for (int e = 0; e < 16; ++e) dst[e] = er[e];
    }
    __syncthreads();

    // ---- phase 2: G = bsum + [x|ret|h] @ [w_ihT; w_hhT]   (4x12 tiles, K=192)
#pragma unroll
    for (int i = 0; i < 6; ++i) {
      int tile = wid * 6 + i;
      int mt = tile / 12, nt = tile % 12;
      int col = nt * 16 + (lane & 15);
      float bb = bsum[col];
      v8f c;
#pragma unroll
      for (int r = 0; r < 8; ++r) c[r] = bb;
#pragma unroll
      for (int kt = 0; kt < 6; ++kt) {
        v16h av = load_a_f16(sA + mt * 16 * ALD + kt * 32, ALD, lane);
        v16h bw = load_b_packed(Gw, kt * 12 + nt, lane);
        c = wmma16(av, bw, c);
      }
      int rbase = mt * 16 + ((lane >> 4) << 3);
#pragma unroll
      for (int r = 0; r < 8; ++r) sG[(rbase + r) * ALD + col] = (_Float16)c[r];
    }
    //       Ghn = bhh_n + h @ w_hhT[:,128:192]   (4x4 tiles, K=64)
#pragma unroll
    for (int i = 0; i < 2; ++i) {
      int tile = wid * 2 + i;
      int mt = tile >> 2, nt = tile & 3;
      int col = nt * 16 + (lane & 15);
      float bb = bhhn[col];
      v8f c;
#pragma unroll
      for (int r = 0; r < 8; ++r) c[r] = bb;
#pragma unroll
      for (int kt = 0; kt < 2; ++kt) {
        v16h av = load_a_f16(sA + mt * 16 * ALD + 128 + kt * 32, ALD, lane);
        v16h bw = load_b_packed(Whn, kt * 4 + nt, lane);
        c = wmma16(av, bw, c);
      }
      int rbase = mt * 16 + ((lane >> 4) << 3);
#pragma unroll
      for (int r = 0; r < 8; ++r) sGhn[(rbase + r) * 64 + col] = (_Float16)c[r];
    }
    __syncthreads();

    // ---- phase 3: gates.  i_n = Gn - Ghn; n = tanh(i_n + r*Ghn)
    for (int i = tid; i < BT * 64; i += NT) {
      int row = i >> 6, j = i & 63;
      float gr = (float)sG[row * ALD + j];
      float gz = (float)sG[row * ALD + 64 + j];
      float gn = (float)sG[row * ALD + 128 + j];
      float hn = (float)sGhn[row * 64 + j];
      float hold = (float)sA[row * ALD + 128 + j];
      float rg = 1.f / (1.f + __expf(-gr));
      float zg = 1.f / (1.f + __expf(-gz));
      float ng = tanhf(gn - hn + rg * hn);
      float hv = (1.f - zg) * ng + zg * hold;
      _Float16 hh = (_Float16)hv;
      sHn[i] = hh;
      sA[row * ALD + 128 + j] = hh;    // next-step h
    }
    __syncthreads();

    // ---- phase 4: logits = h@head_wT + head_b ; q = h@wqT + bq
#pragma unroll
    for (int i = 0; i < 2; ++i) {
      int tile = wid * 2 + i;
      int mt = tile >> 2, nt = tile & 3;
      int col = nt * 16 + (lane & 15);
      int rbase = mt * 16 + ((lane >> 4) << 3);
      {
        float bb = head_b[col];
        v8f c;
#pragma unroll
        for (int r = 0; r < 8; ++r) c[r] = bb;
#pragma unroll
        for (int kt = 0; kt < 2; ++kt) {
          v16h av = load_a_f16(sHn + mt * 16 * 64 + kt * 32, 64, lane);
          v16h bw = load_b_packed(Hw, kt * 4 + nt, lane);
          c = wmma16(av, bw, c);
        }
#pragma unroll
        for (int r = 0; r < 8; ++r) {
          sLg[(rbase + r) * 64 + col] = c[r];
          if (t == L - 1) out[(size_t)(r0 + rbase + r) * VOC + col] = c[r];
        }
      }
      {
        float bb = bq[col];
        v8f c;
#pragma unroll
        for (int r = 0; r < 8; ++r) c[r] = bb;
#pragma unroll
        for (int kt = 0; kt < 2; ++kt) {
          v16h av = load_a_f16(sHn + mt * 16 * 64 + kt * 32, 64, lane);
          v16h bw = load_b_packed(Qw, kt * 4 + nt, lane);
          c = wmma16(av, bw, c);
        }
#pragma unroll
        for (int r = 0; r < 8; ++r) sQ[(rbase + r) * 64 + col] = (_Float16)c[r];
      }
    }
    if (tid == 0) *sEnt = 0.f;
    __syncthreads();

    // ---- phase 5: batch-global entropy gate (atomic sum + counter spin)
    if (tid < BT) {
      const float* lg = sLg + tid * 64;
      float mx = -3.0e38f;
#pragma unroll 8
      for (int j = 0; j < VOC; ++j) mx = fmaxf(mx, lg[j]);
      float S = 0.f, T = 0.f;
#pragma unroll 8
      for (int j = 0; j < VOC; ++j) {
        float d = lg[j] - mx;
        float e = __expf(d);
        S += e; T += d * e;
      }
      atomicAdd(sEnt, __logf(S) - T / S);
    }
    __syncthreads();
    if (tid == 0) {
      atomicAdd(entSum + t, *sEnt);
      __threadfence();
      atomicAdd(entCnt + t, 1u);
      while (atomicAdd(entCnt + t, 0u) < (unsigned)nblk)
        __builtin_amdgcn_s_sleep(1);
      __threadfence();
      float tot = atomicAdd(entSum + t, 0.f);
      *sCond = (tot / (float)B > ENT_THRESH) ? 1.f : 0.f;
    }
    __syncthreads();
    const float cf = *sCond;
    rc += (cf > 0.5f) ? 1 : 0;

    // ---- phase 6: attention against L2-resident f16 K/V (4 lanes per row)
    if (cf > 0.5f && t != L - 1) {
      for (int i = tid; i < BT * 64; i += NT) sAtt[i] = 0.f;
      __syncthreads();
      const int row = tid >> 2, part = tid & 3;
      const int Mq = M >> 2;
      const size_t base = ((size_t)(r0 + row) * M + (size_t)part * Mq) * HD;
      const _Float16* Krow = K16 + base;
      const _Float16* Vrow = V16 + base;
      float qv[64];
#pragma unroll
      for (int j = 0; j < 64; ++j) qv[j] = (float)sQ[row * 64 + j];
      float mx = -3.0e38f;
      for (int m = 0; m < Mq; ++m) {
        const _Float16* kp = Krow + (size_t)m * HD;
        __builtin_prefetch(kp + 4 * HD, 0, 1);
        float s = 0.f;
#pragma unroll
        for (int j = 0; j < 64; ++j) s += qv[j] * (float)kp[j];
        mx = fmaxf(mx, s * ATT_SCALE);
      }
      mx = fmaxf(mx, __shfl_xor(mx, 1, 4));
      mx = fmaxf(mx, __shfl_xor(mx, 2, 4));
      float S = 0.f, acc[64];
#pragma unroll
      for (int j = 0; j < 64; ++j) acc[j] = 0.f;
      for (int m = 0; m < Mq; ++m) {
        const _Float16* kp = Krow + (size_t)m * HD;
        const _Float16* vp = Vrow + (size_t)m * HD;
        __builtin_prefetch(vp + 4 * HD, 0, 1);
        float s = 0.f;
#pragma unroll
        for (int j = 0; j < 64; ++j) s += qv[j] * (float)kp[j];
        float w = __expf(s * ATT_SCALE - mx);
        S += w;
#pragma unroll
        for (int j = 0; j < 64; ++j) acc[j] += w * (float)vp[j];
      }
      S += __shfl_xor(S, 1, 4);
      S += __shfl_xor(S, 2, 4);
      float inv = 1.f / S;
#pragma unroll
      for (int j = 0; j < 64; ++j) atomicAdd(&sAtt[row * 64 + j], acc[j] * inv);
      __syncthreads();
      for (int i = tid; i < BT * 64; i += NT) {
        int r2 = i >> 6, cc = i & 63;
        sA[r2 * ALD + 64 + cc] = (_Float16)sAtt[i];
      }
    } else {
      for (int i = tid; i < BT * 64; i += NT) {
        int r2 = i >> 6, cc = i & 63;
        sA[r2 * ALD + 64 + cc] = (_Float16)0.f;
      }
    }
    __syncthreads();
  }

  if (blockIdx.x == 0 && tid == 0)
    out[(size_t)B * VOC] = (float)rc / (float)L;
}

// ---------------------------------------------------------------------------
extern "C" void kernel_launch(void* const* d_in, const int* in_sizes, int n_in,
                              void* d_out, int out_size, void* d_ws, size_t ws_size,
                              hipStream_t stream) {
  (void)n_in; (void)out_size;
  const int*   seq    = (const int*)  d_in[0];
  const float* memory = (const float*)d_in[1];
  const float* embed  = (const float*)d_in[2];
  const float* w_ih   = (const float*)d_in[3];
  const float* w_hh   = (const float*)d_in[4];
  const float* b_ih   = (const float*)d_in[5];
  const float* b_hh   = (const float*)d_in[6];
  const float* wq     = (const float*)d_in[7];
  const float* bq     = (const float*)d_in[8];
  const float* wk     = (const float*)d_in[9];
  const float* bk     = (const float*)d_in[10];
  const float* wv     = (const float*)d_in[11];
  const float* bv     = (const float*)d_in[12];
  const float* head_w = (const float*)d_in[13];
  const float* head_b = (const float*)d_in[14];
  float* out = (float*)d_out;

  const int B = 1024;
  const int L = in_sizes[0] / B;
  const int M = in_sizes[1] / (B * HD);
  const long BM = (long)B * M;

  char* ws = (char*)d_ws;
  size_t off = 0;
  auto alloc = [&](size_t bytes) {
    off = (off + 255) & ~(size_t)255;
    size_t o = off; off += bytes; return o;
  };
  size_t entSum_o = alloc((size_t)L * 4);
  size_t entCnt_o = alloc((size_t)L * 4);
  size_t Gw_o   = alloc(72 * 512 * 2);
  size_t Whn_o  = alloc(8 * 512 * 2);
  size_t Hw_o   = alloc(8 * 512 * 2);
  size_t Qw_o   = alloc(8 * 512 * 2);
  size_t Kw_o   = alloc(8 * 512 * 2);
  size_t Vw_o   = alloc(8 * 512 * 2);
  size_t bsum_o = alloc(192 * 4);
  size_t bhhn_o = alloc(64 * 4);
  size_t emb_o  = alloc(64 * 64 * 2);
  size_t K16_o  = alloc((size_t)BM * HD * 2);
  size_t V16_o  = alloc((size_t)BM * HD * 2);
  if (off > ws_size) return;  // workspace too small

  float*        entSum = (float*)(ws + entSum_o);
  unsigned int* entCnt = (unsigned int*)(ws + entCnt_o);
  _Float16* Gw   = (_Float16*)(ws + Gw_o);
  _Float16* Whn  = (_Float16*)(ws + Whn_o);
  _Float16* Hw   = (_Float16*)(ws + Hw_o);
  _Float16* Qw   = (_Float16*)(ws + Qw_o);
  _Float16* Kw   = (_Float16*)(ws + Kw_o);
  _Float16* Vw   = (_Float16*)(ws + Vw_o);
  float*    bsum = (float*)(ws + bsum_o);
  float*    bhhn = (float*)(ws + bhhn_o);
  _Float16* embh = (_Float16*)(ws + emb_o);
  _Float16* K16  = (_Float16*)(ws + K16_o);
  _Float16* V16  = (_Float16*)(ws + V16_o);

  init_sync_kernel<<<(L + 255) / 256, 256, 0, stream>>>(entSum, entCnt, L);
  pack_kernel<<<64, NT, 0, stream>>>(w_ih, w_hh, b_ih, b_hh, wq, wk, wv,
                                     head_w, embed, Gw, Whn, Hw, Qw, Kw, Vw,
                                     bsum, bhhn, embh);
  kv_kernel<<<(int)(BM / 64), NT, 0, stream>>>(memory, Kw, Vw, bk, bv, K16, V16);

  (void)hipFuncSetAttribute(reinterpret_cast<const void*>(gru_kernel),
                            hipFuncAttributeMaxDynamicSharedMemorySize,
                            SMEM_BYTES);
  gru_kernel<<<B / BT, NT, SMEM_BYTES, stream>>>(
      seq, K16, V16, Gw, Whn, Hw, Qw, bsum, bhhn, embh, bq, head_b,
      entSum, entCnt, out, B, L, M);
}
